// ScannedRNN_65352222376255
// MI455X (gfx1250) — matrix operand order repaired and checked by reading, not compiled, not run on hardware
//
#include <hip/hip_runtime.h>

#define HIDDEN 512
#define BATCH 128
#define SEQ 512
#define NGATE (3 * HIDDEN)

typedef __attribute__((ext_vector_type(16))) __bf16        bf16x16;
typedef __attribute__((ext_vector_type(8)))  float         f32x8;
typedef __attribute__((ext_vector_type(4)))  unsigned int  u32x4;

union TileBF {
    bf16x16 v;
    u32x4   q[2];
};

__device__ __forceinline__ unsigned short f2bf(float f) {
    // round-to-nearest-even f32 -> bf16
    unsigned int u = __float_as_uint(f);
    u += 0x7FFFu + ((u >> 16) & 1u);
    return (unsigned short)(u >> 16);
}

// sigmoid via hardware v_exp_f32 + v_rcp_f32 (TRANS ops, co-execute with VALU)
__device__ __forceinline__ float fast_sigmoid(float x) {
    return __builtin_amdgcn_rcpf(1.0f + __expf(-x));
}
// tanh(x) = 2*sigmoid(2x) - 1
__device__ __forceinline__ float fast_tanh(float x) {
    return 2.0f * __builtin_amdgcn_rcpf(1.0f + __expf(-2.0f * x)) - 1.0f;
}

// ---------------------------------------------------------------------------
// Phase 0: convert both weight matrices fp32 -> bf16 once (L2-resident after).
// ---------------------------------------------------------------------------
__global__ void convert_weights(const float* __restrict__ wih_f,
                                const float* __restrict__ whh_f,
                                unsigned short* __restrict__ wih_bf,
                                unsigned short* __restrict__ whh_bf) {
    const int n = NGATE * HIDDEN;
    for (int i = blockIdx.x * blockDim.x + threadIdx.x; i < n;
         i += gridDim.x * blockDim.x) {
        wih_bf[i] = f2bf(wih_f[i]);
        whh_bf[i] = f2bf(whh_f[i]);
    }
}

// ---------------------------------------------------------------------------
// Phase 1: persistent GRU scan. 8 workgroups (one per 16-row batch tile),
// 1024 threads = 32 waves; wave w owns hidden columns [16w, 16w+16).
// Per step: fused x@W_ih^T and h@W_hh^T via v_wmma_f32_16x16x32_bf16,
// f32 accumulate, f32 carried hidden state. Weight B-tiles stream from L2
// every step (3 MB bf16, fully L2-resident); K-loop is kept rolled so the
// compiler cannot hoist+spill the weight stream to scratch.
// ---------------------------------------------------------------------------
__global__ __launch_bounds__(1024, 1)
void gru_scan(const float* __restrict__ hidden0,        // [B,H]
              const float* __restrict__ ins,            // [B,T,H]
              const unsigned char* __restrict__ resets, // [B,T] (bool)
              const unsigned short* __restrict__ wih,   // bf16 [3H,H]
              const unsigned short* __restrict__ whh,   // bf16 [3H,H]
              const float* __restrict__ b_ih,           // [3H]
              const float* __restrict__ b_hh,           // [3H]
              float* __restrict__ hfinal,               // [B,H]
              float* __restrict__ ys)                   // [B,T,H]
{
    __shared__ unsigned short lds_x[16 * HIDDEN];        // bf16 x tile
    __shared__ unsigned short lds_hb[2][16 * HIDDEN];    // bf16 h (double buf)
    __shared__ float          lds_hf[16 * HIDDEN];       // f32 carried h

    const int tid  = threadIdx.x;
    const int lane = tid & 31;
    const int wave = tid >> 5;       // 0..31 -> hidden tile
    const int half = lane >> 4;      // 0/1 (lane group)
    const int l15  = lane & 15;
    const int b0   = blockIdx.x * 16;

    // --- init carried hidden state from input ---
    for (int i = tid; i < 16 * HIDDEN; i += 1024) {
        int row = i >> 9, col = i & (HIDDEN - 1);
        float h = hidden0[(b0 + row) * HIDDEN + col];
        lds_hf[i]    = h;
        lds_hb[0][i] = f2bf(h);
    }

    // --- per-lane constants: output column + hoisted biases ---
    const int j0 = wave * 16;
    const int jc = j0 + l15;                    // this lane's output column
    const float bias_r  = b_ih[jc] + b_hh[jc];
    const float bias_z  = b_ih[HIDDEN + jc] + b_hh[HIDDEN + jc];
    const float bias_in = b_ih[2 * HIDDEN + jc];
    const float bias_hn = b_hh[2 * HIDDEN + jc];

    // B-matrix row base: lane (0..15) reads weight row (gate_off + j0 + l15),
    // lanes 16..31 read the same rows at K+16. Contiguous -> coalesced b128.
    const int zoff = HIDDEN * HIDDEN;       // +512 rows
    const int noff = 2 * HIDDEN * HIDDEN;   // +1024 rows
    const int koff_b = half * 16;   // B tile: lanes>=16 hold K 16..31
    const int koff_a = half * 16;   // A tile: lanes>=16 hold K 16..31
    const unsigned short* wih_r = wih + (j0 + l15) * HIDDEN + koff_b;
    const unsigned short* whh_r = whh + (j0 + l15) * HIDDEN + koff_b;

    __syncthreads();

    for (int t = 0; t < SEQ; ++t) {
        const int cur = t & 1, nxt = cur ^ 1;

        // --- phase A: stage x_t as bf16; apply reset mask to carried h ---
        for (int i = tid; i < 16 * HIDDEN; i += 1024) {
            int row = i >> 9, col = i & (HIDDEN - 1);
            size_t gidx = ((size_t)(b0 + row) * SEQ + t) * HIDDEN + col;
            lds_x[i] = f2bf(ins[gidx]);
            if (resets[(b0 + row) * SEQ + t]) {
                lds_hf[i]      = 0.0f;
                lds_hb[cur][i] = 0;
            }
        }
        __syncthreads();

        // --- phase B: fused dual GEMM over K=512 ---
        f32x8 acc_r, acc_z, acc_in, acc_hn;
#pragma unroll
        for (int g = 0; g < 8; ++g) {
            acc_r[g]  = bias_r;
            acc_z[g]  = bias_z;
            acc_in[g] = bias_in;
            acc_hn[g] = bias_hn;
        }

        // Rolled on purpose: addresses depend on k0 so LICM cannot hoist the
        // weight stream across t (which previously caused scratch spilling).
#pragma unroll 1
        for (int k0 = 0; k0 < HIDDEN; k0 += 32) {
            // A tiles (x_t row, h_t row) from LDS, contiguous k = 16*half + s, same order as B
            TileBF ax, ah;
            const unsigned short* xp = &lds_x[l15 * HIDDEN + k0 + koff_a];
            ax.q[0] = *(const u32x4*)(xp);
            ax.q[1] = *(const u32x4*)(xp + 8);
            const unsigned short* hp = &lds_hb[cur][l15 * HIDDEN + k0 + koff_a];
            ah.q[0] = *(const u32x4*)(hp);
            ah.q[1] = *(const u32x4*)(hp + 8);

            // B tiles (weights) straight from L2: 2x b128 per lane per tile
            TileBF br_i, br_h, bz_i, bz_h, bn_i, bn_h;
            const unsigned short* p;
            p = wih_r + k0;        br_i.q[0] = *(const u32x4*)p; br_i.q[1] = *(const u32x4*)(p + 8);
            p = whh_r + k0;        br_h.q[0] = *(const u32x4*)p; br_h.q[1] = *(const u32x4*)(p + 8);
            p = wih_r + zoff + k0; bz_i.q[0] = *(const u32x4*)p; bz_i.q[1] = *(const u32x4*)(p + 8);
            p = whh_r + zoff + k0; bz_h.q[0] = *(const u32x4*)p; bz_h.q[1] = *(const u32x4*)(p + 8);
            p = wih_r + noff + k0; bn_i.q[0] = *(const u32x4*)p; bn_i.q[1] = *(const u32x4*)(p + 8);
            p = whh_r + noff + k0; bn_h.q[0] = *(const u32x4*)p; bn_h.q[1] = *(const u32x4*)(p + 8);

            acc_r  = __builtin_amdgcn_wmma_f32_16x16x32_bf16(false, ax.v, false, br_i.v, (short)0, acc_r,  false, false);
            acc_r  = __builtin_amdgcn_wmma_f32_16x16x32_bf16(false, ah.v, false, br_h.v, (short)0, acc_r,  false, false);
            acc_z  = __builtin_amdgcn_wmma_f32_16x16x32_bf16(false, ax.v, false, bz_i.v, (short)0, acc_z,  false, false);
            acc_z  = __builtin_amdgcn_wmma_f32_16x16x32_bf16(false, ah.v, false, bz_h.v, (short)0, acc_z,  false, false);
            acc_in = __builtin_amdgcn_wmma_f32_16x16x32_bf16(false, ax.v, false, bn_i.v, (short)0, acc_in, false, false);
            acc_hn = __builtin_amdgcn_wmma_f32_16x16x32_bf16(false, ah.v, false, bn_h.v, (short)0, acc_hn, false, false);
        }

        // --- epilogue: gates + state update; C/D layout row = g + 8*half ---
#pragma unroll
        for (int g = 0; g < 8; ++g) {
            const int row = g + half * 8;
            const float r = fast_sigmoid(acc_r[g]);
            const float z = fast_sigmoid(acc_z[g]);
            const float n = fast_tanh(acc_in[g] + r * acc_hn[g]);
            const float hold = lds_hf[row * HIDDEN + jc];
            const float hnew = (1.0f - z) * n + z * hold;
            ys[((size_t)(b0 + row) * SEQ + t) * HIDDEN + jc] = hnew;
            lds_hf[row * HIDDEN + jc]      = hnew;
            lds_hb[nxt][row * HIDDEN + jc] = f2bf(hnew);
        }
        __syncthreads();
    }

    // --- final hidden state ---
    for (int i = tid; i < 16 * HIDDEN; i += 1024) {
        int row = i >> 9, col = i & (HIDDEN - 1);
        hfinal[(b0 + row) * HIDDEN + col] = lds_hf[i];
    }
}

// ---------------------------------------------------------------------------
// Harness entry. Inputs (dict order): hidden, ins, resets, W_ih, W_hh,
// b_ih, b_hh. Output: h_final [B,H] then ys [B,T,H], concatenated flat.
// ---------------------------------------------------------------------------
extern "C" void kernel_launch(void* const* d_in, const int* in_sizes, int n_in,
                              void* d_out, int out_size, void* d_ws, size_t ws_size,
                              hipStream_t stream) {
    const float*         hidden0 = (const float*)d_in[0];
    const float*         ins     = (const float*)d_in[1];
    const unsigned char* resets  = (const unsigned char*)d_in[2];
    const float*         wih_f   = (const float*)d_in[3];
    const float*         whh_f   = (const float*)d_in[4];
    const float*         b_ih    = (const float*)d_in[5];
    const float*         b_hh    = (const float*)d_in[6];

    // Workspace: bf16 weights (3 MB total)
    unsigned short* wih_bf = (unsigned short*)d_ws;
    unsigned short* whh_bf = wih_bf + (size_t)NGATE * HIDDEN;

    float* out    = (float*)d_out;
    float* hfinal = out;                            // [B,H]
    float* ys     = out + (size_t)BATCH * HIDDEN;   // [B,T,H]

    convert_weights<<<1024, 256, 0, stream>>>(wih_f, whh_f, wih_bf, whh_bf);
    gru_scan<<<BATCH / 16, 1024, 0, stream>>>(hidden0, ins, resets,
                                              wih_bf, whh_bf, b_ih, b_hh,
                                              hfinal, ys);
}
